// BraakAwareAttention_18769007083634
// MI455X (gfx1250) — matrix operand order repaired and checked by reading, not compiled, not run on hardware
//
#include <hip/hip_runtime.h>

#define BATCH   8
#define SEQ     1024
#define DIM     1024
#define NSTAGES 6

typedef __attribute__((ext_vector_type(16))) __bf16       v16bf;
typedef __attribute__((ext_vector_type(8)))  float        v8f;
typedef __attribute__((ext_vector_type(4)))  unsigned int v4u;
typedef __attribute__((ext_vector_type(4)))  int          v4i;
typedef __attribute__((ext_vector_type(8)))  int          v8i;

// Dynamic-LDS layout for attn_main (dynamic LDS base == 0: no static __shared__)
#define OFF_LOGITS 0        // 16*1024 f32  (64 KB)
#define OFF_PBF    65536    // 16*1024 bf16 (32 KB)
#define OFF_QLDS   98304    // 16*1024 bf16 (32 KB)
#define OFF_RED    131072   // 16*16 f32
#define OFF_RMAX   132096   // 16 f32
#define OFF_RSUM   132160   // 16 f32
#define SMEM_BYTES 132224

union FragB16 { v16bf v; unsigned int u[8]; };

__device__ __forceinline__ unsigned short f2bf(float f) {
    unsigned int u = __float_as_uint(f);
    u += 0x7fffu + ((u >> 16) & 1u);          // round-to-nearest-even
    return (unsigned short)(u >> 16);
}

// A-fragment, 16-bit 16x32 (ISA 7.12.2): lane&15 = M; lane>=16 shifts K by 8.
__device__ __forceinline__ v16bf load_a16(const unsigned short* base, int ld,
                                          int row0, int k0) {
    const int lane = threadIdx.x & 31;
    const int kh   = lane >> 4;
    const unsigned short* p = base + (size_t)(row0 + (lane & 15)) * ld + k0 + 8 * kh;
    FragB16 f;
#pragma unroll
    for (int j = 0; j < 4; ++j) {
        f.u[j]     = *(const unsigned int*)(p + 2 * j);        // K = 2j, 2j+1
        f.u[j + 4] = *(const unsigned int*)(p + 2 * j + 16);   // K = 16+2j, 17+2j
    }
    return f.v;
}

// B-fragment, 16-bit 32x16, source stored [N][K] row-major:
// lane&15 = N ; element e<8 -> K = e + 8*kh, e>=8 -> K = 16 + 8*kh + (e-8)
__device__ __forceinline__ v16bf load_b16(const unsigned short* base, int ld,
                                          int n0, int k0) {
    const int lane = threadIdx.x & 31;
    const int kh   = lane >> 4;
    const unsigned short* p = base + (size_t)(n0 + (lane & 15)) * ld + k0 + 8 * kh;
    FragB16 f;
#pragma unroll
    for (int j = 0; j < 8; ++j)
        f.u[j] = *(const unsigned int*)(p + 2 * j + 8 * (j >> 2));
    return f.v;
}

// Tensor Data Mover: 1-D copy of `nelem` bf16 elements (contiguous) into LDS.
// D# group0: count/lds_addr/global_addr/type ; group1: sizes & strides.
__device__ __forceinline__ void tdm_load_1d_bf16(const unsigned short* gsrc,
                                                 unsigned lds_off, unsigned nelem) {
    unsigned long long ga = (unsigned long long)gsrc;
    v4u g0;
    g0.x = 1u;                                                  // count=1 (valid), user mode
    g0.y = lds_off;                                             // lds_addr
    g0.z = (unsigned)(ga & 0xFFFFFFFFu);                        // global_addr[31:0]
    g0.w = (unsigned)((ga >> 32) & 0x01FFFFFFu) | 0x80000000u;  // addr[56:32] | type=2
    v8i g1;
    g1[0] = (int)(1u << 16);                          // workgroup_mask=0, data_size=1 (2B)
    g1[1] = (int)((nelem & 0xFFFFu) << 16);           // tensor_dim0[15:0] @ bits[63:48]
    g1[2] = (int)(((nelem >> 16) & 0xFFFFu)           // tensor_dim0[31:16]
                  | (1u << 16));                      // tensor_dim1 = 1 (low half)
    g1[3] = (int)((nelem & 0xFFFFu) << 16);           // tile_dim0 @ bits[127:112]
    g1[4] = 1;                                        // tile_dim1 = 1, tile_dim2 = 0
    g1[5] = (int)nelem;                               // tensor_dim0_stride[31:0]
    g1[6] = 0;                                        // stride hi, dim1_stride lo
    g1[7] = 0;
    v4i gz = {0, 0, 0, 0};
#if __clang_major__ >= 23
    v8i gz8 = {0, 0, 0, 0, 0, 0, 0, 0};
    __builtin_amdgcn_tensor_load_to_lds(g0, g1, gz, gz, gz8, 0);
#else
    __builtin_amdgcn_tensor_load_to_lds(g0, g1, gz, gz, 0);
#endif
}

// ---------------- prep: Q+bias -> bf16, K -> bf16 ----------------
__global__ __launch_bounds__(256) void prep_qk(
    const float* __restrict__ query, const float* __restrict__ key,
    const float* __restrict__ embed, const int* __restrict__ stages,
    unsigned short* __restrict__ Qb, unsigned short* __restrict__ Kb)
{
    const int bs = blockIdx.x;            // b * SEQ + s
    const int b  = bs / SEQ;
    const int s  = bs % SEQ;
    int st = stages[b];
    if (st < 0) st = 0;
    if (st >= NSTAGES) st = NSTAGES - 1;
    const float bias = embed[st * DIM + s];   // bias broadcasts on s, not d
    const size_t base = (size_t)bs * DIM;
    for (int d = threadIdx.x; d < DIM; d += 256) {
        Qb[base + d] = f2bf(query[base + d] + bias);
        Kb[base + d] = f2bf(key[base + d]);
    }
}

// ------- prep: V -> Vt[b][d][t] bf16, async global->LDS staging -------
__global__ __launch_bounds__(256) void prep_v(
    const float* __restrict__ value, unsigned short* __restrict__ Vt)
{
    __shared__ float tile[32][33];
    const int b  = blockIdx.z;
    const int t0 = blockIdx.x * 32;
    const int d0 = blockIdx.y * 32;
    const int x  = threadIdx.x & 31;
    const int y  = threadIdx.x >> 5;
    const float* vp = value + (size_t)b * SEQ * DIM;
#pragma unroll
    for (int r = y; r < 32; r += 8) {
        // LDS offset of tile[r][x]; static shared base is 0 (only LDS object)
        unsigned     ldsoff = (unsigned)((r * 33 + x) * 4);
        const float* g      = vp + (size_t)(t0 + r) * DIM + d0 + x;
        asm volatile("global_load_async_to_lds_b32 %0, %1, off"
                     :: "v"(ldsoff), "v"(g) : "memory");
    }
    asm volatile("s_wait_asynccnt 0x0" ::: "memory");
    __syncthreads();
    unsigned short* op = Vt + (size_t)b * DIM * SEQ;
#pragma unroll
    for (int r = y; r < 32; r += 8)
        op[(size_t)(d0 + r) * SEQ + t0 + x] = f2bf(tile[x][r]);
}

// ---------------- fused attention: 16 S-rows per block ----------------
__global__ __launch_bounds__(256) void attn_main(
    const unsigned short* __restrict__ Qb,
    const unsigned short* __restrict__ Kb,
    const unsigned short* __restrict__ Vt,
    float* __restrict__ out)
{
    extern __shared__ char smem[];
    float*          logits = (float*)(smem + OFF_LOGITS);
    unsigned short* pbf    = (unsigned short*)(smem + OFF_PBF);
    const unsigned short* qlds = (const unsigned short*)(smem + OFF_QLDS);
    float*          red    = (float*)(smem + OFF_RED);
    float*          rowmax = (float*)(smem + OFF_RMAX);
    float*          rowsum = (float*)(smem + OFF_RSUM);

    const int b    = blockIdx.y;
    const int s0   = blockIdx.x * 16;
    const int tid  = threadIdx.x;
    const int wave = tid >> 5;
    const int lane = tid & 31;
    const int kh   = lane >> 4;
    const int lmod = lane & 15;

    const unsigned short* Qp = Qb + (size_t)b * SEQ * DIM;
    const unsigned short* Kp = Kb + (size_t)b * SEQ * DIM;
    const unsigned short* Vp = Vt + (size_t)b * DIM * SEQ;

    // ---- TDM: stage the contiguous 16x1024 bf16 Q tile into LDS ----
    if (wave == 0) {
        tdm_load_1d_bf16(Qp + (size_t)s0 * DIM, OFF_QLDS, 16 * 1024);
        __builtin_amdgcn_s_wait_tensorcnt(0);
    }
    __syncthreads();

    // ---- Phase A: logits = (Q+bias) * K^T ; each wave owns 8 N-tiles ----
    {
        v8f acc[8] = {};
        for (int kt = 0; kt < DIM / 32; ++kt) {
            v16bf a = load_a16(qlds, DIM, 0, kt * 32);       // A from LDS
            if (kt + 1 < DIM / 32)
                __builtin_prefetch(Kp + (size_t)(wave * 128 + lmod) * DIM + (kt + 1) * 32, 0, 0);
#pragma unroll
            for (int i = 0; i < 8; ++i) {
                const int nt = wave * 8 + i;
                v16bf bb = load_b16(Kp, DIM, nt * 16, kt * 32);
                acc[i] = __builtin_amdgcn_wmma_f32_16x16x32_bf16(
                    false, a, false, bb, (short)0, acc[i], false, false);
            }
        }
#pragma unroll
        for (int i = 0; i < 8; ++i) {
            const int nt = wave * 8 + i;
#pragma unroll
            for (int j = 0; j < 8; ++j)
                logits[(j + 8 * kh) * SEQ + nt * 16 + lmod] = acc[i][j];
        }
    }
    __syncthreads();

    // ---- softmax over 16 rows x 1024 cols, 16 threads per row ----
    {
        const int r  = tid >> 4;
        const int c0 = tid & 15;
        float m = -3.4e38f;
        for (int c = c0; c < SEQ; c += 16) m = fmaxf(m, logits[r * SEQ + c]);
        red[r * 16 + c0] = m;
        __syncthreads();
        if (c0 == 0) {
            float mm = red[r * 16];
#pragma unroll
            for (int i = 1; i < 16; ++i) mm = fmaxf(mm, red[r * 16 + i]);
            rowmax[r] = mm;
        }
        __syncthreads();
        const float mrow = rowmax[r];
        float s = 0.f;
        for (int c = c0; c < SEQ; c += 16) {
            float e = __expf(logits[r * SEQ + c] - mrow);
            logits[r * SEQ + c] = e;
            s += e;
        }
        red[r * 16 + c0] = s;
        __syncthreads();
        if (c0 == 0) {
            float ss = 0.f;
#pragma unroll
            for (int i = 0; i < 16; ++i) ss += red[r * 16 + i];
            rowsum[r] = ss;
        }
        __syncthreads();
        const float inv = 1.0f / rowsum[r];
        for (int c = c0; c < SEQ; c += 16)
            pbf[r * SEQ + c] = f2bf(logits[r * SEQ + c] * inv);
    }
    __syncthreads();

    // ---- Phase B: out = P * V ; each wave owns 8 D-tiles ----
    {
        v8f acc[8] = {};
        for (int kt = 0; kt < SEQ / 32; ++kt) {
            v16bf a = load_a16(pbf, SEQ, 0, kt * 32);        // A from LDS
            if (kt + 1 < SEQ / 32)
                __builtin_prefetch(Vp + (size_t)(wave * 128 + lmod) * SEQ + (kt + 1) * 32, 0, 0);
#pragma unroll
            for (int i = 0; i < 8; ++i) {
                const int dt = wave * 8 + i;
                v16bf bb = load_b16(Vp, SEQ, dt * 16, kt * 32);
                acc[i] = __builtin_amdgcn_wmma_f32_16x16x32_bf16(
                    false, a, false, bb, (short)0, acc[i], false, false);
            }
        }
        float* op = out + ((size_t)b * SEQ + s0) * DIM;
#pragma unroll
        for (int i = 0; i < 8; ++i) {
            const int dt = wave * 8 + i;
#pragma unroll
            for (int j = 0; j < 8; ++j)
                op[(size_t)(j + 8 * kh) * DIM + dt * 16 + lmod] = acc[i][j];
        }
    }
}

extern "C" void kernel_launch(void* const* d_in, const int* in_sizes, int n_in,
                              void* d_out, int out_size, void* d_ws, size_t ws_size,
                              hipStream_t stream) {
    const float* query  = (const float*)d_in[0];
    const float* key    = (const float*)d_in[1];
    const float* value  = (const float*)d_in[2];
    const float* embed  = (const float*)d_in[3];
    const int*   stages = (const int*)d_in[4];
    float* out = (float*)d_out;

    unsigned short* Qb = (unsigned short*)d_ws;                 // 16 MB
    unsigned short* Kb = Qb + (size_t)BATCH * SEQ * DIM;        // 16 MB
    unsigned short* Vt = Kb + (size_t)BATCH * SEQ * DIM;        // 16 MB

    (void)hipFuncSetAttribute((const void*)attn_main,
                              hipFuncAttributeMaxDynamicSharedMemorySize, SMEM_BYTES);

    prep_qk<<<dim3(BATCH * SEQ), 256, 0, stream>>>(query, key, embed, stages, Qb, Kb);
    prep_v<<<dim3(SEQ / 32, DIM / 32, BATCH), 256, 0, stream>>>(value, Vt);
    attn_main<<<dim3(SEQ / 16, BATCH), 256, SMEM_BYTES, stream>>>(Qb, Kb, Vt, out);
}
